// VisionEncoderLoss_42039139893253
// MI455X (gfx1250) — matrix operand order repaired and checked by reading, not compiled, not run on hardware
//
#include <hip/hip_runtime.h>
#include <math.h>

// ---------------------------------------------------------------------------
// Types
// ---------------------------------------------------------------------------
typedef __bf16 bf16;
typedef __attribute__((ext_vector_type(16))) __bf16 v16bf;
typedef __attribute__((ext_vector_type(8)))  float  v8f;
typedef __attribute__((ext_vector_type(4)))  unsigned u32x4;
typedef __attribute__((ext_vector_type(8)))  int      i32x8;
typedef __attribute__((ext_vector_type(4)))  int      i32x4;

struct alignas(16) F4  { float x, y, z, w; };
struct alignas(16) BF8 { bf16 h[8]; };
union  V16 { BF8 b[2]; v16bf v; };

#define NIMG 128
#define NS   576
#define NT   1024
#define DS_  1024
#define DT_  1152

__device__ inline unsigned pk2(float a, float b) {
    union { bf16 h[2]; unsigned u; } x;
    x.h[0] = (bf16)a; x.h[1] = (bf16)b;
    return x.u;
}

__device__ inline float wred_sum(float a) {
    for (int o = 16; o; o >>= 1) a += __shfl_xor(a, o);
    return a;
}

// LDS byte offset of a __shared__ object (flat shared addr keeps offset in low 32b)
__device__ inline unsigned lds_off(const void* p) {
    return (unsigned)(unsigned long long)p;
}

// ---------------------------------------------------------------------------
// TDM: load a 128-row x 32-col bf16 tile (row stride = stride_elems) into LDS
// with 16B padding after each 64B row => LDS row stride 40 bf16 elements.
// Issued by a single wave (caller guards); tracked with TENSORcnt.
// ---------------------------------------------------------------------------
__device__ inline void tdm_load_tile_b(const bf16* gptr, unsigned lds_byte,
                                       int stride_elems) {
    unsigned long long ga = (unsigned long long)gptr;   // byte address
    u32x4 g0;
    g0[0] = 1u;                                          // count=1, user desc
    g0[1] = lds_byte;                                    // lds_addr
    g0[2] = (unsigned)ga;                                // global_addr[31:0]
    g0[3] = (unsigned)((ga >> 32) & 0x01FFFFFFu)         // global_addr[56:32]
          | (2u << 30);                                  // type=2 ("image")
    i32x8 g1;
    g1[0] = (1 << 16)       // data_size = 1 (2 bytes)
          | (1 << 20)       // pad_enable
          | (3 << 22)       // pad_interval: 16 DWORDs (64B) between pads
          | (3 << 25);      // pad_amount: 4 DWORDs (16B)
    g1[1] = 0;                                           // no atomic barrier; dim0 lo16=0
    g1[2] = 0x10;                                        // tensor_dim0 = 1<<20 (hi16), dim1 lo16=0
    g1[3] = 0x10 | (32 << 16);                           // tensor_dim1 = 1<<20; tile_dim0 = 32
    g1[4] = 128;                                         // tile_dim1 = 128 rows; tile_dim2 = 0
    g1[5] = stride_elems;                                // tensor_dim0_stride lo32 (elements)
    g1[6] = 0;                                           // stride0 hi16 | stride1 lo16
    g1[7] = 0;
    i32x4 z4 = {0, 0, 0, 0};
    i32x8 z8 = {0, 0, 0, 0, 0, 0, 0, 0};
    __builtin_amdgcn_tensor_load_to_lds(g0, g1, z4, z4, z8, 0);
}

// A fragment (16x32 bf16, M x K): lane m = l&15, khalf = l>>4
__device__ inline v16bf load_frag_a(const bf16* lds, int r0, int stride) {
    int lane = threadIdx.x & 31;
    const bf16* p = lds + (r0 + (lane & 15)) * stride + ((lane >> 4) << 3);
    V16 x;
    x.b[0] = *(const BF8*)(p);
    x.b[1] = *(const BF8*)(p + 16);
    return x.v;
}

// B fragment (32x16 bf16, K x N) from n-major LDS tile [n][k]
__device__ inline v16bf load_frag_b(const bf16* lds, int n0, int stride) {
    int lane = threadIdx.x & 31;
    const bf16* p = lds + (n0 + (lane & 15)) * stride + ((lane >> 4) << 3);
    V16 x;
    x.b[0] = *(const BF8*)(p);
    x.b[1] = *(const BF8*)(p + 16);
    return x.v;
}

// ---------------------------------------------------------------------------
// K0: init accumulators + per-image c2 min/max (ordered-uint float bits)
// ---------------------------------------------------------------------------
__global__ void vel_init(unsigned* mnb, unsigned* mxb, float* acc) {
    int t = threadIdx.x;
    if (t < NIMG) { mnb[t] = 0x7F800000u; mxb[t] = 0u; }
    if (t < 2) acc[t] = 0.0f;
}

// ---------------------------------------------------------------------------
// K1: Wt[n][k] = bf16(W[k][n])  (32x32 LDS transpose tiles)
// ---------------------------------------------------------------------------
__global__ __launch_bounds__(256) void vel_convert_w(const float* __restrict__ W,
                                                     bf16* __restrict__ Wt) {
    __shared__ float tile[32][33];
    int n0 = blockIdx.x * 32, k0 = blockIdx.y * 32;
    int t = threadIdx.x;
    int r = t >> 3, c = (t & 7) * 4;
    F4 f = *(const F4*)(W + (size_t)(k0 + r) * DS_ + n0 + c);
    tile[r][c] = f.x; tile[r][c + 1] = f.y; tile[r][c + 2] = f.z; tile[r][c + 3] = f.w;
    __syncthreads();
    unsigned* dst = (unsigned*)(Wt + (size_t)(n0 + r) * DT_ + k0 + c);
    dst[0] = pk2(tile[c][r],     tile[c + 1][r]);
    dst[1] = pk2(tile[c + 2][r], tile[c + 3][r]);
}

// ---------------------------------------------------------------------------
// K2: proj_bf16[img] = bf16(tea[img] @ W + b)   -- 128x128 block tile, K=1152
//     A: fp32->bf16 register-pipelined through LDS; B: TDM double-buffered
// ---------------------------------------------------------------------------
__global__ __launch_bounds__(256) void vel_proj_gemm(const float* __restrict__ teaq,
                                                     const float* __restrict__ teap,
                                                     const bf16* __restrict__ Wt,
                                                     const float* __restrict__ bias,
                                                     bf16* __restrict__ proj) {
    __shared__ __align__(16) bf16 shA[128 * 40];
    __shared__ __align__(16) bf16 shB[2][128 * 40];
    int img = blockIdx.z;
    const float* tea = (img < 64) ? teaq + (size_t)img * NT * DT_
                                  : teap + (size_t)(img - 64) * NT * DT_;
    int m_base = blockIdx.y * 128, n_base = blockIdx.x * 128;
    int t = threadIdx.x, wid = t >> 5, lane = t & 31;
    int wm = wid & 1, wn = wid >> 1;
    int uw = __builtin_amdgcn_readfirstlane(wid);

    int ra = t >> 3, ca = (t & 7) * 4;                  // A global/LDS coords
    const bf16* Bbase = Wt + (size_t)n_base * DT_;      // row n_base, k=0

    const int NK = DT_ / 32;
    F4 areg[4];
    for (int p = 0; p < 4; p++)
        areg[p] = *(const F4*)(tea + (size_t)(m_base + ra + p * 32) * DT_ + ca);
    if (uw == 0) tdm_load_tile_b(Bbase, lds_off(&shB[0][0]), DT_);

    v8f acc[4][2] = {};
    int buf = 0;
    for (int s = 0; s < NK; ++s) {
        for (int p = 0; p < 4; p++) {                   // publish A(s)
            unsigned* d = (unsigned*)&shA[(ra + p * 32) * 40 + ca];
            d[0] = pk2(areg[p].x, areg[p].y);
            d[1] = pk2(areg[p].z, areg[p].w);
        }
        if (uw == 0) {
            if (s + 1 < NK) {                           // prefetch B(s+1), wait B(s)
                tdm_load_tile_b(Bbase + (s + 1) * 32, lds_off(&shB[buf ^ 1][0]), DT_);
                __builtin_amdgcn_s_wait_tensorcnt(1);
            } else {
                __builtin_amdgcn_s_wait_tensorcnt(0);
            }
        }
        __syncthreads();
        if (s + 1 < NK) {                               // prefetch A(s+1) into regs
            int kk = (s + 1) * 32;
            for (int p = 0; p < 4; p++)
                areg[p] = *(const F4*)(tea + (size_t)(m_base + ra + p * 32) * DT_ + kk + ca);
        }
        v16bf fa[4], fb[2];
        const bf16* sb = shB[buf];
        for (int mi = 0; mi < 4; mi++) fa[mi] = load_frag_a(shA, wm * 64 + mi * 16, 40);
        for (int ni = 0; ni < 2; ni++) fb[ni] = load_frag_b(sb, wn * 32 + ni * 16, 40);
        for (int mi = 0; mi < 4; mi++)
            for (int ni = 0; ni < 2; ni++)
                acc[mi][ni] = __builtin_amdgcn_wmma_f32_16x16x32_bf16(
                    false, fa[mi], false, fb[ni], (short)0, acc[mi][ni], false, false);
        __syncthreads();
        buf ^= 1;
    }

    int khalf = lane >> 4, nl = lane & 15;
    for (int mi = 0; mi < 4; mi++)
        for (int ni = 0; ni < 2; ni++)
            for (int v = 0; v < 8; v++) {
                int m = m_base + wm * 64 + mi * 16 + v + (khalf << 3);
                int n = n_base + wn * 32 + ni * 16 + nl;
                float val = acc[mi][ni][v] + bias[n];
                proj[((size_t)img << 20) + ((size_t)m << 10) + n] = (bf16)val;
            }
}

// ---------------------------------------------------------------------------
// K3a: np[row] = sum_k proj_bf16[row][k]^2
// ---------------------------------------------------------------------------
__global__ __launch_bounds__(256) void vel_rownorm_proj(const bf16* __restrict__ proj,
                                                        float* __restrict__ np) {
    int wid = threadIdx.x >> 5, lane = threadIdx.x & 31;
    size_t row = (size_t)blockIdx.x * 8 + wid;
    const bf16* p = proj + row * DS_ + lane * 32;
    float a = 0.f;
    for (int c = 0; c < 4; c++) {
        BF8 x = *(const BF8*)(p + c * 8);
        for (int e = 0; e < 8; e++) { float f = (float)x.h[e]; a += f * f; }
    }
    a = wred_sum(a);
    if (lane == 0) np[row] = a;
}

// ---------------------------------------------------------------------------
// K3b: ns[row] = sum_k bf16(stu[row][k])^2
// ---------------------------------------------------------------------------
__global__ __launch_bounds__(256) void vel_rownorm_stu(const float* __restrict__ stuq,
                                                       const float* __restrict__ stup,
                                                       float* __restrict__ ns) {
    int wid = threadIdx.x >> 5, lane = threadIdx.x & 31;
    int row = blockIdx.x * 8 + wid;
    int img = row / NS, i = row % NS;
    const float* base = ((img < 64) ? stuq + (size_t)img * NS * DS_
                                    : stup + (size_t)(img - 64) * NS * DS_) + (size_t)i * DS_;
    const float* p = base + lane * 32;
    float a = 0.f;
    for (int c = 0; c < 8; c++) {
        F4 f = *(const F4*)(p + c * 4);
        float v0 = (float)(bf16)f.x, v1 = (float)(bf16)f.y;
        float v2 = (float)(bf16)f.z, v3 = (float)(bf16)f.w;
        a += v0 * v0 + v1 * v1 + v2 * v2 + v3 * v3;
    }
    a = wred_sum(a);
    if (lane == 0) ns[row] = a;
}

// ---------------------------------------------------------------------------
// K4: sq = max(ns+np-2*(stu@projT), 0) stored f32; per-image min/max of sqrt(sq)
//     64x128 block tile, K=1024; A register-pipelined, B via TDM double buffer
// ---------------------------------------------------------------------------
__global__ __launch_bounds__(256) void vel_c2_gemm(const float* __restrict__ stuq,
                                                   const float* __restrict__ stup,
                                                   const bf16* __restrict__ proj,
                                                   const float* __restrict__ ns,
                                                   const float* __restrict__ np,
                                                   float* __restrict__ c2sq,
                                                   unsigned* __restrict__ mnb,
                                                   unsigned* __restrict__ mxb) {
    __shared__ __align__(16) bf16 shA[64 * 40];
    __shared__ __align__(16) bf16 shB[2][128 * 40];
    __shared__ float s_ns[64], s_np[128], smin[8], smax[8];
    int img = blockIdx.z;
    const float* stu = (img < 64) ? stuq + (size_t)img * NS * DS_
                                  : stup + (size_t)(img - 64) * NS * DS_;
    const bf16* pimg = proj + ((size_t)img << 20);
    int m_base = blockIdx.y * 64, n_base = blockIdx.x * 128;
    int t = threadIdx.x, wid = t >> 5, lane = t & 31;
    int wm = wid & 1, wn = wid >> 1;
    int uw = __builtin_amdgcn_readfirstlane(wid);

    if (t < 64)           s_ns[t]      = ns[img * NS + m_base + t];
    else if (t < 192)     s_np[t - 64] = np[img * NT + n_base + (t - 64)];

    int ra = t >> 3, ca = (t & 7) * 4;
    const bf16* Bbase = pimg + (size_t)n_base * DS_;

    const int NK = DS_ / 32;
    F4 areg[2];
    for (int p = 0; p < 2; p++)
        areg[p] = *(const F4*)(stu + (size_t)(m_base + ra + p * 32) * DS_ + ca);
    if (uw == 0) tdm_load_tile_b(Bbase, lds_off(&shB[0][0]), DS_);

    v8f acc[2][2] = {};
    int buf = 0;
    for (int s = 0; s < NK; ++s) {
        for (int p = 0; p < 2; p++) {
            unsigned* d = (unsigned*)&shA[(ra + p * 32) * 40 + ca];
            d[0] = pk2(areg[p].x, areg[p].y);
            d[1] = pk2(areg[p].z, areg[p].w);
        }
        if (uw == 0) {
            if (s + 1 < NK) {
                tdm_load_tile_b(Bbase + (s + 1) * 32, lds_off(&shB[buf ^ 1][0]), DS_);
                __builtin_amdgcn_s_wait_tensorcnt(1);
            } else {
                __builtin_amdgcn_s_wait_tensorcnt(0);
            }
        }
        __syncthreads();
        if (s + 1 < NK) {
            int kk = (s + 1) * 32;
            for (int p = 0; p < 2; p++)
                areg[p] = *(const F4*)(stu + (size_t)(m_base + ra + p * 32) * DS_ + kk + ca);
        }
        v16bf fa[2], fb[2];
        const bf16* sb = shB[buf];
        for (int mi = 0; mi < 2; mi++) fa[mi] = load_frag_a(shA, wm * 32 + mi * 16, 40);
        for (int ni = 0; ni < 2; ni++) fb[ni] = load_frag_b(sb, wn * 32 + ni * 16, 40);
        for (int mi = 0; mi < 2; mi++)
            for (int ni = 0; ni < 2; ni++)
                acc[mi][ni] = __builtin_amdgcn_wmma_f32_16x16x32_bf16(
                    false, fa[mi], false, fb[ni], (short)0, acc[mi][ni], false, false);
        __syncthreads();
        buf ^= 1;
    }

    int khalf = lane >> 4, nl = lane & 15;
    float lmin = 1e30f, lmax = 0.0f;
    for (int mi = 0; mi < 2; mi++)
        for (int ni = 0; ni < 2; ni++)
            for (int v = 0; v < 8; v++) {
                int m = wm * 32 + mi * 16 + v + (khalf << 3);
                int n = wn * 32 + ni * 16 + nl;
                float sq = s_ns[m] + s_np[n] - 2.0f * acc[mi][ni][v];
                sq = fmaxf(sq, 0.0f);
                float c2 = sqrtf(sq);
                c2sq[((size_t)img * NS + m_base + m) * NT + n_base + n] = sq;
                lmin = fminf(lmin, c2);
                lmax = fmaxf(lmax, c2);
            }
    for (int o = 16; o; o >>= 1) {
        lmin = fminf(lmin, __shfl_xor(lmin, o));
        lmax = fmaxf(lmax, __shfl_xor(lmax, o));
    }
    if (lane == 0) { smin[wid] = lmin; smax[wid] = lmax; }
    __syncthreads();
    if (t == 0) {
        float bmn = smin[0], bmx = smax[0];
        for (int w = 1; w < 8; w++) { bmn = fminf(bmn, smin[w]); bmx = fmaxf(bmx, smax[w]); }
        atomicMin(&mnb[img], __float_as_uint(bmn));   // valid: c2 >= 0
        atomicMax(&mxb[img], __float_as_uint(bmx));
    }
}

// ---------------------------------------------------------------------------
// K5: per-row argmin of c1 + 1e-3*c2norm; accumulate sq at argmin (== MSE term)
// ---------------------------------------------------------------------------
__global__ __launch_bounds__(256) void vel_argmin(const float* __restrict__ c2sq,
                                                  const unsigned* __restrict__ mnb,
                                                  const unsigned* __restrict__ mxb,
                                                  float* __restrict__ acc,
                                                  float c1min, float c1scale, float kdnorm) {
    int wid = threadIdx.x >> 5, lane = threadIdx.x & 31;
    int row = blockIdx.x * 8 + wid;
    int img = row / NS, i = row % NS;
    float c2mn = __uint_as_float(mnb[img]);
    float c2mx = __uint_as_float(mxb[img]);
    float c2sc = 1e-3f / (c2mx - c2mn + 1e-6f);
    int iy = i / 24, ix = i % 24;
    float sy = (iy + 0.5f) * (2.0f / 24.0f) - 1.0f;
    float sx = (ix + 0.5f) * (2.0f / 24.0f) - 1.0f;
    const float* prow = c2sq + (size_t)row * NT;

    float bestc = 1e30f, bestsq = 0.0f;
    int bestj = 0;
    for (int j = lane; j < NT; j += 32) {
        float sq = prow[j];
        float c2 = sqrtf(sq);
        int jy = j >> 5, jx = j & 31;
        float ty = (jy + 0.5f) * 0.0625f - 1.0f;
        float tx = (jx + 0.5f) * 0.0625f - 1.0f;
        float c1 = (fabsf(sy - ty) + fabsf(sx - tx) - c1min) * c1scale;
        float cost = c1 + (c2 - c2mn) * c2sc;
        if (cost < bestc) { bestc = cost; bestj = j; bestsq = sq; }
    }
    for (int o = 16; o; o >>= 1) {
        float oc  = __shfl_xor(bestc, o);
        int   oj  = __shfl_xor(bestj, o);
        float osq = __shfl_xor(bestsq, o);
        if (oc < bestc || (oc == bestc && oj < bestj)) { bestc = oc; bestj = oj; bestsq = osq; }
    }
    if (lane == 0) atomicAdd(&acc[0], bestsq * kdnorm);
}

// ---------------------------------------------------------------------------
// K6: contrastive loss (fp32, one block per query row)
// ---------------------------------------------------------------------------
__global__ __launch_bounds__(256) void vel_contrastive(const float* __restrict__ qry,
                                                       const float* __restrict__ pos,
                                                       float* __restrict__ acc) {
    __shared__ float part[256];
    __shared__ float sc[64];
    int i = blockIdx.x, t = threadIdx.x;
    int j = t >> 2, s = t & 3;
    const float* q = qry + (size_t)i * DS_;
    const float* p = pos + (size_t)j * DS_;
    float a = 0.f;
    for (int d = s; d < DS_; d += 4) a += q[d] * p[d];
    part[t] = a;
    __syncthreads();
    if (s == 0) sc[j] = (part[t] + part[t + 1] + part[t + 2] + part[t + 3]) * 50.0f; // 1/TEMP
    __syncthreads();
    if (t == 0) {
        float mx = sc[0];
        for (int k = 1; k < 64; k++) mx = fmaxf(mx, sc[k]);
        float se = 0.f;
        for (int k = 0; k < 64; k++) se += expf(sc[k] - mx);
        float logp = sc[i] - mx - logf(se);
        atomicAdd(acc, -logp * (1.0f / 64.0f));
    }
}

// ---------------------------------------------------------------------------
// K7: out = contrastive + kd
// ---------------------------------------------------------------------------
__global__ void vel_finalize(const float* __restrict__ acc, float* __restrict__ out) {
    out[0] = acc[0] + acc[1];
}

// ---------------------------------------------------------------------------
// Host: c1 range is input-independent (separable manhattan grid distance)
// ---------------------------------------------------------------------------
static void c1_range_host(float& c1min, float& c1scale) {
    double mn = 1e300, mx = -1e300;
    for (int i = 0; i < 24; i++)
        for (int j = 0; j < 32; j++) {
            double s = (i + 0.5) / 24.0 * 2.0 - 1.0;
            double tt = (j + 0.5) / 32.0 * 2.0 - 1.0;
            double d = fabs(s - tt);
            if (d < mn) mn = d;
            if (d > mx) mx = d;
        }
    double lo = 2.0 * mn, hi = 2.0 * mx;   // y and x ranges are identical
    c1min = (float)lo;
    c1scale = (float)(1.0 / (hi - lo + 1e-6));
}

extern "C" void kernel_launch(void* const* d_in, const int* in_sizes, int n_in,
                              void* d_out, int out_size, void* d_ws, size_t ws_size,
                              hipStream_t stream) {
    const float* stu_q = (const float*)d_in[0];
    const float* tea_q = (const float*)d_in[1];
    const float* stu_p = (const float*)d_in[2];
    const float* tea_p = (const float*)d_in[3];
    const float* qry   = (const float*)d_in[4];
    const float* pos   = (const float*)d_in[5];
    const float* W     = (const float*)d_in[6];
    const float* bias  = (const float*)d_in[7];

    char* ws = (char*)d_ws;
    // workspace layout (~574 MB)
    bf16*     proj = (bf16*)(ws);                          // 128*1024*1024*2 = 268435456
    float*    c2sq = (float*)(ws + 268435456ULL);          // 128*576*1024*4  = 301989888
    bf16*     Wt   = (bf16*)(ws + 570425344ULL);           // 1024*1152*2     = 2359296
    float*    np   = (float*)(ws + 572784640ULL);          // 128*1024*4      = 524288
    float*    ns   = (float*)(ws + 573308928ULL);          // 128*576*4       = 294912
    unsigned* mnb  = (unsigned*)(ws + 573603840ULL);       // 512
    unsigned* mxb  = (unsigned*)(ws + 573604352ULL);       // 512
    float*    acc  = (float*)(ws + 573604864ULL);          // [0]=kd, [1]=contrastive

    float c1min, c1scale;
    c1_range_host(c1min, c1scale);
    float kdnorm = (float)(1.0 / ((2.0 * 64.0 + 1e-8) * 576.0 * 1024.0));

    vel_init<<<1, 256, 0, stream>>>(mnb, mxb, acc);
    vel_convert_w<<<dim3(32, 36), 256, 0, stream>>>(W, Wt);
    vel_proj_gemm<<<dim3(8, 8, NIMG), 256, 0, stream>>>(tea_q, tea_p, Wt, bias, proj);
    vel_rownorm_proj<<<(NIMG * NT) / 8, 256, 0, stream>>>(proj, np);
    vel_rownorm_stu<<<(NIMG * NS) / 8, 256, 0, stream>>>(stu_q, stu_p, ns);
    vel_c2_gemm<<<dim3(8, 9, NIMG), 256, 0, stream>>>(stu_q, stu_p, proj, ns, np, c2sq, mnb, mxb);
    vel_argmin<<<(NIMG * NS) / 8, 256, 0, stream>>>(c2sq, mnb, mxb, acc, c1min, c1scale, kdnorm);
    vel_contrastive<<<64, 256, 0, stream>>>(qry, pos, acc + 1);
    vel_finalize<<<1, 1, 0, stream>>>(acc, (float*)d_out);
}
